// BalancedMoELayer_40982577938614
// MI455X (gfx1250) — matrix operand order repaired and checked by reading, not compiled, not run on hardware
//
#include <hip/hip_runtime.h>
#include <hip/hip_bf16.h>

// ---------------------------------------------------------------------------
// MoE (top-2 of 8 experts), T=4096 tokens, H=1024, I=2048.
// bf16 WMMA (v_wmma_f32_16x16x32_bf16) with fp32 accumulation.
// 4 concurrent N-tiles per wave -> 4 independent WMMA dependency chains.
// ---------------------------------------------------------------------------

#define T_TOK 4096
#define HDIM  1024
#define IDIM  2048
#define NEXP  8
#define TM    16                  // tokens per tile
#define XS_STRIDE (HDIM + 8)      // padded LDS strides (bank-conflict free b128)
#define HS_STRIDE (IDIM + 8)

typedef __attribute__((ext_vector_type(16))) __bf16 bf16x16;
typedef __attribute__((ext_vector_type(8)))  __bf16 bf16x8;
typedef __attribute__((ext_vector_type(8)))  float  f32x8;

union Frag { bf16x16 v; bf16x8 h[2]; };

__device__ __forceinline__ float fast_silu(float v) {
    // v * sigmoid(v) with v_exp_f32 + v_rcp_f32 (no IEEE divide expansion)
    return v * __builtin_amdgcn_rcpf(1.0f + __expf(-v));
}

// ---------------------------------------------------------------------------
// Router: one wave32 per token. logits = x @ gate_w^T, softmax, top-2.
// ---------------------------------------------------------------------------
__global__ void router_kernel(const float* __restrict__ x,
                              const float* __restrict__ gate_w,
                              int* __restrict__ top_idx,   // [T*2]
                              float* __restrict__ top_w,   // [T*2]
                              int* __restrict__ counts)    // [NEXP]
{
    const int wave = threadIdx.x >> 5;
    const int lane = threadIdx.x & 31;
    const int t = blockIdx.x * 8 + wave;
    if (t >= T_TOK) return;

    float acc[NEXP];
#pragma unroll
    for (int e = 0; e < NEXP; ++e) acc[e] = 0.0f;

    const float* xr = x + (size_t)t * HDIM;
    for (int h = lane; h < HDIM; h += 32) {
        float xv = xr[h];
#pragma unroll
        for (int e = 0; e < NEXP; ++e)
            acc[e] += xv * gate_w[e * HDIM + h];
    }
#pragma unroll
    for (int s = 16; s >= 1; s >>= 1) {
#pragma unroll
        for (int e = 0; e < NEXP; ++e)
            acc[e] += __shfl_xor(acc[e], s, 32);
    }
    if (lane == 0) {
        float m = acc[0];
#pragma unroll
        for (int e = 1; e < NEXP; ++e) m = fmaxf(m, acc[e]);
        float p[NEXP]; float sum = 0.0f;
#pragma unroll
        for (int e = 0; e < NEXP; ++e) { p[e] = __expf(acc[e] - m); sum += p[e]; }
        float inv = 1.0f / sum;
        float v0 = -1.0f, v1 = -1.0f; int i0 = 0, i1 = 0;
#pragma unroll
        for (int e = 0; e < NEXP; ++e) {
            float pe = p[e] * inv;
            if (pe > v0)      { v1 = v0; i1 = i0; v0 = pe; i0 = e; }
            else if (pe > v1) { v1 = pe; i1 = e; }
        }
        float winv = 1.0f / (v0 + v1 + 1e-9f);
        top_idx[t * 2 + 0] = i0;  top_w[t * 2 + 0] = v0 * winv;
        top_idx[t * 2 + 1] = i1;  top_w[t * 2 + 1] = v1 * winv;
        atomicAdd(&counts[i0], 1);
        atomicAdd(&counts[i1], 1);
    }
}

// ---------------------------------------------------------------------------
// Prefix offsets (padded to TM) + init slot arrays + reset cursors.
// ---------------------------------------------------------------------------
__global__ void offsets_kernel(const int* __restrict__ counts,
                               int* __restrict__ offsets,   // [NEXP+1]
                               int* __restrict__ cursors,   // [NEXP]
                               int* __restrict__ slot_tok,
                               float* __restrict__ slot_w,
                               int slot_cap)
{
    const int tid = threadIdx.x;
    if (tid == 0) {
        int off = 0;
        for (int e = 0; e < NEXP; ++e) {
            offsets[e] = off;
            int padded = ((counts[e] + TM - 1) / TM) * TM;
            off += padded;
            cursors[e] = 0;
        }
        offsets[NEXP] = off;
    }
    for (int i = tid; i < slot_cap; i += blockDim.x) {
        slot_tok[i] = -1;
        slot_w[i] = 0.0f;
    }
}

__global__ void scatter_kernel(const int* __restrict__ top_idx,
                               const float* __restrict__ top_w,
                               const int* __restrict__ offsets,
                               int* __restrict__ cursors,
                               int* __restrict__ slot_tok,
                               float* __restrict__ slot_w)
{
    int t = blockIdx.x * blockDim.x + threadIdx.x;
    if (t >= T_TOK) return;
#pragma unroll
    for (int k = 0; k < 2; ++k) {
        int e = top_idx[t * 2 + k];
        int pos = atomicAdd(&cursors[e], 1);
        int s = offsets[e] + pos;
        slot_tok[s] = t;
        slot_w[s] = top_w[t * 2 + k];
    }
}

// ---------------------------------------------------------------------------
// Tiled fp32 -> bf16 transpose: src[e][R][C] -> dst[e][C][R].
// ---------------------------------------------------------------------------
__global__ void transpose_bf16_kernel(const float* __restrict__ src,
                                      __bf16* __restrict__ dst,
                                      int R, int C)
{
    __shared__ float tile[32][33];
    const int e  = blockIdx.z;
    const int c0 = blockIdx.x * 32;
    const int r0 = blockIdx.y * 32;
    const float* s = src + (size_t)e * R * C;
    __bf16*      d = dst + (size_t)e * R * C;
    const int tx = threadIdx.x;   // 0..31
    const int ty = threadIdx.y;   // 0..7
#pragma unroll
    for (int i = 0; i < 32; i += 8)
        tile[ty + i][tx] = s[(size_t)(r0 + ty + i) * C + (c0 + tx)];
    __syncthreads();
#pragma unroll
    for (int i = 0; i < 32; i += 8)
        d[(size_t)(c0 + ty + i) * R + (r0 + tx)] = (__bf16)tile[tx][ty + i];
}

// ---------------------------------------------------------------------------
// Main grouped dual-GEMM. One block = one (expert, 16-token tile).
// 256 threads = 8 waves. WMMA bf16 16x16x32, f32 accum.
// Each wave processes 4 N-tiles concurrently: one shared A fragment,
// 4 B fragments, 4 accumulators (independent WMMA chains).
//   A fragment (16x32 bf16): lane m=lane&15, half=lane>>4:
//     elems 0..7  = A[m][kk + half*16 + 0..7]
//     elems 8..15 = A[m][kk + half*16 + 8..15]
//   B fragment (32x16 bf16): lane n=lane&15, half=lane>>4:
//     elems j     = B[kk + half*16 + j][n]  (contiguous K in transposed weights)
//   C/D (16x16 f32): elem r -> row half*8+r, col n.
// ---------------------------------------------------------------------------
__global__ void moe_expert_gemm(const float* __restrict__ x,      // [T,H]
                                const __bf16* __restrict__ wupT,  // [E][I][H]
                                const __bf16* __restrict__ wdnT,  // [E][H][I]
                                const int* __restrict__ counts,   // [E]
                                const int* __restrict__ offsets,  // [E+1]
                                const int* __restrict__ slot_tok,
                                const float* __restrict__ slot_w,
                                float* __restrict__ out)          // [T,H]
{
    extern __shared__ char smem[];
    __bf16* Xs  = (__bf16*)smem;                                     // TM*XS_STRIDE
    __bf16* Hs  = (__bf16*)(smem + (size_t)TM * XS_STRIDE * 2);      // TM*HS_STRIDE
    int*    tok = (int*)  (smem + (size_t)TM * XS_STRIDE * 2 + (size_t)TM * HS_STRIDE * 2);
    float*  wgt = (float*)(tok + TM);

    const int e     = blockIdx.y;
    const int cnt   = counts[e];
    const int tile0 = blockIdx.x * TM;
    if (tile0 >= cnt) return;
    const int base = offsets[e] + tile0;

    const int tid  = threadIdx.x;
    const int wave = tid >> 5;
    const int lane = tid & 31;
    const int n    = lane & 15;
    const int half = lane >> 4;

    if (tid < TM) {
        tok[tid] = slot_tok[base + tid];
        wgt[tid] = slot_w[base + tid];
    }
    __syncthreads();

    // Gather X rows (fp32 -> bf16) into LDS.
    for (int idx = tid; idx < TM * HDIM; idx += 256) {
        int r = idx >> 10;            // / HDIM
        int c = idx & (HDIM - 1);
        int t = tok[r];
        float v = (t >= 0) ? x[(size_t)t * HDIM + c] : 0.0f;
        Xs[r * XS_STRIDE + c] = (__bf16)v;
    }
    __syncthreads();

    // ---- GEMM1: Hs[t][i] = silu(sum_h Xs[t][h] * WupT[e][i][h]) ----
    {
        const __bf16* wup_e = wupT + (size_t)e * IDIM * HDIM;
        const __bf16* arow  = Xs + n * XS_STRIDE;
        // 128 tiles over 8 waves = 16 per wave, processed 4 at a time.
#pragma unroll
        for (int g = 0; g < 4; ++g) {
            int n0[4];
            const __bf16* bcol[4];
            f32x8 acc[4];
#pragma unroll
            for (int j = 0; j < 4; ++j) {
                n0[j]   = (wave + 8 * (g * 4 + j)) * 16;
                bcol[j] = wup_e + (size_t)(n0[j] + n) * HDIM;
                acc[j]  = 0.0f;
            }
#pragma unroll 2
            for (int kk = 0; kk < HDIM; kk += 32) {
                Frag a;
                const __bf16* ar = arow + kk + half * 16;
                a.h[0] = *(const bf16x8*)(ar);
                a.h[1] = *(const bf16x8*)(ar + 8);
#pragma unroll
                for (int j = 0; j < 4; ++j) {
                    Frag b;
                    const __bf16* br = bcol[j] + kk + half * 16;
                    b.h[0] = *(const bf16x8*)(br);
                    b.h[1] = *(const bf16x8*)(br + 8);
                    acc[j] = __builtin_amdgcn_wmma_f32_16x16x32_bf16(
                        false, a.v, false, b.v, (short)0, acc[j], false, false);
                }
            }
#pragma unroll
            for (int j = 0; j < 4; ++j) {
#pragma unroll
                for (int r = 0; r < 8; ++r) {
                    Hs[(half * 8 + r) * HS_STRIDE + n0[j] + n] =
                        (__bf16)fast_silu(acc[j][r]);
                }
            }
        }
    }
    __syncthreads();

    // ---- GEMM2: out[t][h] += wgt[t] * sum_i Hs[t][i] * WdnT[e][h][i] ----
    {
        const __bf16* wdn_e = wdnT + (size_t)e * HDIM * IDIM;
        const __bf16* arow  = Hs + n * HS_STRIDE;
        // 64 tiles over 8 waves = 8 per wave, processed 4 at a time.
#pragma unroll
        for (int g = 0; g < 2; ++g) {
            int h0[4];
            const __bf16* bcol[4];
            f32x8 acc[4];
#pragma unroll
            for (int j = 0; j < 4; ++j) {
                h0[j]   = (wave + 8 * (g * 4 + j)) * 16;
                bcol[j] = wdn_e + (size_t)(h0[j] + n) * IDIM;
                acc[j]  = 0.0f;
            }
#pragma unroll 2
            for (int kk = 0; kk < IDIM; kk += 32) {
                Frag a;
                const __bf16* ar = arow + kk + half * 16;
                a.h[0] = *(const bf16x8*)(ar);
                a.h[1] = *(const bf16x8*)(ar + 8);
#pragma unroll
                for (int j = 0; j < 4; ++j) {
                    Frag b;
                    const __bf16* br = bcol[j] + kk + half * 16;
                    b.h[0] = *(const bf16x8*)(br);
                    b.h[1] = *(const bf16x8*)(br + 8);
                    acc[j] = __builtin_amdgcn_wmma_f32_16x16x32_bf16(
                        false, a.v, false, b.v, (short)0, acc[j], false, false);
                }
            }
#pragma unroll
            for (int r = 0; r < 8; ++r) {
                int row = half * 8 + r;
                int t = tok[row];
                if (t >= 0) {
                    float w = wgt[row];
#pragma unroll
                    for (int j = 0; j < 4; ++j)
                        atomicAdd(out + (size_t)t * HDIM + h0[j] + n, w * acc[j][r]);
                }
            }
        }
    }
}

// ---------------------------------------------------------------------------
// Host launch
// ---------------------------------------------------------------------------
static inline size_t align_up(size_t v, size_t a) { return (v + a - 1) & ~(a - 1); }

extern "C" void kernel_launch(void* const* d_in, const int* in_sizes, int n_in,
                              void* d_out, int out_size, void* d_ws, size_t ws_size,
                              hipStream_t stream) {
    (void)in_sizes; (void)n_in; (void)out_size; (void)ws_size;
    const float* x      = (const float*)d_in[0];  // [B,S,H] = [T,H]
    const float* gate_w = (const float*)d_in[1];  // [E,H]
    const float* w_up   = (const float*)d_in[2];  // [E,H,I]
    const float* w_down = (const float*)d_in[3];  // [E,I,H]
    float* out = (float*)d_out;                   // [T,H]

    const int slot_cap = 2 * T_TOK + NEXP * TM;   // padded slot capacity

    // Workspace carve-out
    char* ws = (char*)d_ws;
    size_t off = 0;
    __bf16* wupT = (__bf16*)(ws + off); off = align_up(off + (size_t)NEXP * IDIM * HDIM * 2, 256);
    __bf16* wdnT = (__bf16*)(ws + off); off = align_up(off + (size_t)NEXP * HDIM * IDIM * 2, 256);
    int*    top_idx = (int*)  (ws + off); off = align_up(off + (size_t)T_TOK * 2 * 4, 256);
    float*  top_w   = (float*)(ws + off); off = align_up(off + (size_t)T_TOK * 2 * 4, 256);
    int*    counts  = (int*)  (ws + off); off = align_up(off + NEXP * 4, 256);
    int*    cursors = (int*)  (ws + off); off = align_up(off + NEXP * 4, 256);
    int*    offsets = (int*)  (ws + off); off = align_up(off + (NEXP + 1) * 4, 256);
    int*    slot_tok = (int*)  (ws + off); off = align_up(off + (size_t)slot_cap * 4, 256);
    float*  slot_w   = (float*)(ws + off); off = align_up(off + (size_t)slot_cap * 4, 256);

    // 0) zero routing counters and output accumulator
    hipMemsetAsync(counts, 0, NEXP * sizeof(int), stream);
    hipMemsetAsync(out, 0, (size_t)T_TOK * HDIM * sizeof(float), stream);

    // 1) one-time weight transposes to bf16 (L2-resident afterwards)
    {
        dim3 blk(32, 8, 1);
        dim3 g_up(IDIM / 32, HDIM / 32, NEXP);    // w_up [H][I] -> wupT [I][H]
        transpose_bf16_kernel<<<g_up, blk, 0, stream>>>(w_up, wupT, HDIM, IDIM);
        dim3 g_dn(HDIM / 32, IDIM / 32, NEXP);    // w_down [I][H] -> wdnT [H][I]
        transpose_bf16_kernel<<<g_dn, blk, 0, stream>>>(w_down, wdnT, IDIM, HDIM);
    }

    // 2) router (one wave32 per token)
    router_kernel<<<T_TOK / 8, 256, 0, stream>>>(x, gate_w, top_idx, top_w, counts);

    // 3) padded prefix offsets + slot init
    offsets_kernel<<<1, 256, 0, stream>>>(counts, offsets, cursors,
                                          slot_tok, slot_w, slot_cap);

    // 4) scatter tokens into expert buckets
    scatter_kernel<<<T_TOK / 256, 256, 0, stream>>>(top_idx, top_w, offsets, cursors,
                                                    slot_tok, slot_w);

    // 5) grouped dual-GEMM (WMMA bf16). Max tiles per expert = 4096/16 = 256.
    const size_t smem_bytes = (size_t)TM * XS_STRIDE * 2   // Xs
                            + (size_t)TM * HS_STRIDE * 2   // Hs
                            + TM * 4 + TM * 4;             // tok + wgt
    hipFuncSetAttribute((const void*)moe_expert_gemm,
                        hipFuncAttributeMaxDynamicSharedMemorySize,
                        (int)smem_bytes);
    dim3 grid(T_TOK / TM, NEXP, 1);
    moe_expert_gemm<<<grid, 256, smem_bytes, stream>>>(
        x, wupT, wdnT, counts, offsets, slot_tok, slot_w, out);
}
